// GCNModelVAE_77970836291823
// MI455X (gfx1250) — matrix operand order repaired and checked by reading, not compiled, not run on hardware
//
#include <hip/hip_runtime.h>

// ---------------- problem constants (match reference) ----------------
namespace cfg {
constexpr int N  = 4096;
constexpr int T  = 2048;
constexpr int D  = 256;
constexpr int H1 = 128;
constexpr int H2 = 32;
constexpr int NH = 4;
constexpr float ALPHA = 0.2f;
}

// ---------------- WMMA types ----------------
typedef __attribute__((ext_vector_type(16))) __bf16         v16bf;
typedef __attribute__((ext_vector_type(2)))  __bf16         v2bf;
typedef __attribute__((ext_vector_type(16))) unsigned short v16u;
typedef __attribute__((ext_vector_type(8)))  unsigned short v8u;
typedef __attribute__((ext_vector_type(4)))  unsigned short v4u;
typedef __attribute__((ext_vector_type(2)))  unsigned int   v2ui;
typedef __attribute__((ext_vector_type(8)))  float          v8f;

// fp32 -> bf16 through the native convert path (backend legalizes to RNE
// integer ops if no packed cvt exists, so this is never worse than manual).
static __device__ __forceinline__ unsigned short f2bf(float x) {
    __bf16 h = (__bf16)x;
    return __builtin_bit_cast(unsigned short, h);
}
// pair convert: lets the backend select packed v_cvt_pk_bf16_f32
static __device__ __forceinline__ unsigned int f2bf2(float x, float y) {
    v2bf p = { (__bf16)x, (__bf16)y };
    return __builtin_bit_cast(unsigned int, p);
}
static __device__ __forceinline__ v4u pack4(float4 f) {
    v2ui p = { f2bf2(f.x, f.y), f2bf2(f.z, f.w) };
    return __builtin_bit_cast(v4u, p);
}

#define BM 128     // fast-kernel block rows
#define BN 64      // block cols
#define BKK 32     // K tile
#define LDSS 40    // BKK + 8 ushorts pad; keeps 8B/16B fragment accesses aligned
#define SM 64      // safe-kernel block rows

// ==================================================================
// FAST GEMM: requires M%128==0, Nc%64==0, K%32==0 (all large GEMMs
// here). 8 waves, each owns a 32x32 tile (2x2 WMMA, 4 accumulators):
// 8 ds_load_b128 feed 4 v_wmma per K-step. float4 global fetch,
// packed LDS stores, double-buffered LDS pipeline, no bounds checks.
// TA: A stored [K,M]; TB: B stored [Nc,K]. EPI: 0 none, 1 relu, 2 elu.
// ==================================================================
template <bool TA, bool TB, int EPI>
__global__ __launch_bounds__(256)
void gemm_wmma_fast(const float* __restrict__ A, int lda,
                    const float* __restrict__ B, int ldb,
                    float* __restrict__ C, int ldc,
                    int K, const float* __restrict__ bias)
{
    __shared__ __align__(16) unsigned short As[2][BM][LDSS];
    __shared__ __align__(16) unsigned short Bs[2][BN][LDSS];   // column-major: Bs[n][k]

    const int tid  = threadIdx.x;
    const int wave = tid >> 5;
    const int lane = tid & 31;
    const int m0 = blockIdx.y * BM;
    const int n0 = blockIdx.x * BN;

    const int wr = wave & 3;   // 32-row tile within block (4 of them)
    const int wc = wave >> 2;  // 32-col tile within block (2 of them)
    const int g  = lane >> 4;
    const int l15 = lane & 15;

    v8f acc00 = {}, acc01 = {}, acc10 = {}, acc11 = {};
    const int nk = K / BKK;

    float4 fa[4], fb[2];

    auto fetch = [&](int kt) {
        const int k0 = kt * BKK;
#pragma unroll
        for (int r = 0; r < 4; ++r) {                   // A: 128x32 tile
            const int idx = tid + r * 256;              // 0..1023
            if (!TA) {                                  // contiguous along K
                int row = idx >> 3, col = (idx & 7) * 4;
                fa[r] = *(const float4*)(A + (size_t)(m0 + row) * lda + (k0 + col));
            } else {                                    // contiguous along M
                int kk = idx >> 5, mm = (idx & 31) * 4;
                fa[r] = *(const float4*)(A + (size_t)(k0 + kk) * lda + (m0 + mm));
            }
        }
#pragma unroll
        for (int r = 0; r < 2; ++r) {                   // B: 32x64 tile
            const int idx = tid + r * 256;              // 0..511
            if (!TB) {                                  // contiguous along N
                int kk = idx >> 4, nn = (idx & 15) * 4;
                fb[r] = *(const float4*)(B + (size_t)(k0 + kk) * ldb + (n0 + nn));
            } else {                                    // contiguous along K
                int nn = idx >> 3, kk = (idx & 7) * 4;
                fb[r] = *(const float4*)(B + (size_t)(n0 + nn) * ldb + (k0 + kk));
            }
        }
    };

    auto stage = [&](int buf) {
#pragma unroll
        for (int r = 0; r < 4; ++r) {
            const int idx = tid + r * 256;
            if (!TA) {
                int row = idx >> 3, col = (idx & 7) * 4;
                *(v4u*)&As[buf][row][col] = pack4(fa[r]);   // 8B ds_store
            } else {
                int kk = idx >> 5, mm = (idx & 31) * 4;
                As[buf][mm + 0][kk] = f2bf(fa[r].x);
                As[buf][mm + 1][kk] = f2bf(fa[r].y);
                As[buf][mm + 2][kk] = f2bf(fa[r].z);
                As[buf][mm + 3][kk] = f2bf(fa[r].w);
            }
        }
#pragma unroll
        for (int r = 0; r < 2; ++r) {
            const int idx = tid + r * 256;
            if (!TB) {
                int kk = idx >> 4, nn = (idx & 15) * 4;
                Bs[buf][nn + 0][kk] = f2bf(fb[r].x);
                Bs[buf][nn + 1][kk] = f2bf(fb[r].y);
                Bs[buf][nn + 2][kk] = f2bf(fb[r].z);
                Bs[buf][nn + 3][kk] = f2bf(fb[r].w);
            } else {
                int nn = idx >> 3, kk = (idx & 7) * 4;
                *(v4u*)&Bs[buf][nn][kk] = pack4(fb[r]);     // 8B ds_store
            }
        }
    };

    // A fragment: ISA 16-bit A layout (chunked K per lane-half)
    auto loadA = [&](int buf, int mrow) -> v16bf {
        v8u lo = *(const v8u*)&As[buf][mrow][8 * g];        // K = 8g..8g+7
        v8u hi = *(const v8u*)&As[buf][mrow][16 + 8 * g];   // K = 16+8g..
        v16u u;
#pragma unroll
        for (int e = 0; e < 8; ++e) { u[e] = lo[e]; u[e + 8] = hi[e]; }
        return __builtin_bit_cast(v16bf, u);
    };
    // B fragment: same chunked K layout as A (K = 8g..8g+7, 16+8g..16+8g+7)
    auto loadB = [&](int buf, int nrow) -> v16bf {
        v8u lo = *(const v8u*)&Bs[buf][nrow][8 * g];
        v8u hi = *(const v8u*)&Bs[buf][nrow][16 + 8 * g];
        v16u u;
#pragma unroll
        for (int e = 0; e < 8; ++e) { u[e] = lo[e]; u[e + 8] = hi[e]; }
        return __builtin_bit_cast(v16bf, u);
    };

    auto compute = [&](int buf) {
        v16bf a0 = loadA(buf, wr * 32 + l15);
        v16bf a1 = loadA(buf, wr * 32 + 16 + l15);
        v16bf b0 = loadB(buf, wc * 32 + l15);
        v16bf b1 = loadB(buf, wc * 32 + 16 + l15);
        acc00 = __builtin_amdgcn_wmma_f32_16x16x32_bf16(false, a0, false, b0, (short)0, acc00, false, false);
        acc01 = __builtin_amdgcn_wmma_f32_16x16x32_bf16(false, a0, false, b1, (short)0, acc01, false, false);
        acc10 = __builtin_amdgcn_wmma_f32_16x16x32_bf16(false, a1, false, b0, (short)0, acc10, false, false);
        acc11 = __builtin_amdgcn_wmma_f32_16x16x32_bf16(false, a1, false, b1, (short)0, acc11, false, false);
    };

    // -------- pipelined main loop: fetch(k+1) overlaps compute(k) --------
    // stage(buf) consumes fa/fb before fetch(kt+1) overwrites them, so no
    // shadow registers / v_mov copies are needed.
    fetch(0);
    int buf = 0;
    for (int kt = 0; kt < nk; ++kt) {
        stage(buf);
        __syncthreads();
        if (kt + 1 < nk) fetch(kt + 1);
        compute(buf);
        buf ^= 1;
    }

    // -------- epilogue (full tiles, no guards) --------
#pragma unroll
    for (int i = 0; i < 2; ++i) {
#pragma unroll
        for (int j = 0; j < 2; ++j) {
            v8f acc = (i == 0) ? (j == 0 ? acc00 : acc01) : (j == 0 ? acc10 : acc11);
            const int ncol = n0 + wc * 32 + j * 16 + l15;
            const float bv = bias ? bias[ncol] : 0.f;
            const int mbase = m0 + wr * 32 + i * 16 + g * 8;
#pragma unroll
            for (int v = 0; v < 8; ++v) {
                float x = acc[v] + bv;
                if (EPI == 1) x = x > 0.f ? x : 0.f;
                if (EPI == 2) x = x > 0.f ? x : (__expf(x) - 1.f);
                C[(size_t)(mbase + v) * ldc + ncol] = x;
            }
        }
    }
}

// ==================================================================
// SAFE GEMM (bounds-checked, non-transposed) for the Nc=32 GEMMs.
// ==================================================================
template <int EPI>
__global__ __launch_bounds__(256)
void gemm_wmma_safe(const float* __restrict__ A, int lda,
                    const float* __restrict__ B, int ldb,
                    float* __restrict__ C, int ldc,
                    int M, int Nc, int K,
                    const float* __restrict__ bias)
{
    __shared__ __align__(16) unsigned short As[SM][LDSS];
    __shared__ __align__(16) unsigned short Bs[BN][LDSS];

    const int tid  = threadIdx.x;
    const int wave = tid >> 5;
    const int lane = tid & 31;
    const int m0 = blockIdx.y * SM;
    const int n0 = blockIdx.x * BN;
    const int wr = wave & 3, wc = wave >> 2;
    const int g    = lane >> 4;
    const int mloc = wr * 16 + (lane & 15);

    v8f acc0 = {};
    v8f acc1 = {};

    for (int k0 = 0; k0 < K; k0 += BKK) {
#pragma unroll
        for (int r = 0; r < 8; ++r) {
            int idx = tid + r * 256;
            int row = idx >> 5, col = idx & 31;
            int gm = m0 + row, gk = k0 + col;
            float v = 0.f;
            if (gm < M && gk < K) v = A[(size_t)gm * lda + gk];
            As[row][col] = f2bf(v);
        }
#pragma unroll
        for (int r = 0; r < 8; ++r) {
            int idx = tid + r * 256;
            int nn = idx >> 5, kk = idx & 31;
            int gn = n0 + nn, gk = k0 + kk;
            float v = 0.f;
            if (gn < Nc && gk < K) v = B[(size_t)gk * ldb + gn];
            Bs[nn][kk] = f2bf(v);
        }
        __syncthreads();

        v16u au;
        {
            v8u lo = *(const v8u*)&As[mloc][8 * g];
            v8u hi = *(const v8u*)&As[mloc][16 + 8 * g];
#pragma unroll
            for (int e = 0; e < 8; ++e) { au[e] = lo[e]; au[e + 8] = hi[e]; }
        }
        v16bf a = __builtin_bit_cast(v16bf, au);
#pragma unroll
        for (int t = 0; t < 2; ++t) {
            const int nloc = wc * 32 + t * 16 + (lane & 15);
            v16u bu;
            {
                v8u lo = *(const v8u*)&Bs[nloc][8 * g];
                v8u hi = *(const v8u*)&Bs[nloc][16 + 8 * g];
#pragma unroll
                for (int e = 0; e < 8; ++e) { bu[e] = lo[e]; bu[e + 8] = hi[e]; }
            }
            v16bf b = __builtin_bit_cast(v16bf, bu);
            if (t == 0)
                acc0 = __builtin_amdgcn_wmma_f32_16x16x32_bf16(false, a, false, b,
                                                               (short)0, acc0, false, false);
            else
                acc1 = __builtin_amdgcn_wmma_f32_16x16x32_bf16(false, a, false, b,
                                                               (short)0, acc1, false, false);
        }
        __syncthreads();
    }

#pragma unroll
    for (int t = 0; t < 2; ++t) {
        v8f acc = (t == 0) ? acc0 : acc1;
        int ncol = n0 + wc * 32 + t * 16 + (lane & 15);
        if (ncol >= Nc) continue;
        float bv = bias ? bias[ncol] : 0.f;
#pragma unroll
        for (int v = 0; v < 8; ++v) {
            int mrow = m0 + wr * 16 + g * 8 + v;
            if (mrow >= M) continue;
            float x = acc[v] + bv;
            if (EPI == 1) x = x > 0.f ? x : 0.f;
            if (EPI == 2) x = x > 0.f ? x : (__expf(x) - 1.f);
            C[(size_t)mrow * ldc + ncol] = x;
        }
    }
}

// ---------------- f/g projections: f[h,n] = Wh[h,n,:]·a_src[h,:] ----------------
__global__ __launch_bounds__(256)
void fg_kernel(const float* __restrict__ Wh, const float* __restrict__ asrc,
               const float* __restrict__ adst, float* __restrict__ f,
               float* __restrict__ g, int n, int Dd, int nh)
{
    int wave = threadIdx.x >> 5, lane = threadIdx.x & 31;
    int row = blockIdx.x * 8 + wave;          // row = h*n + i
    if (row >= nh * n) return;                // wave-uniform
    int h = row / n;
    const float* w  = Wh   + (size_t)row * Dd;
    const float* as = asrc + (size_t)h * Dd;
    const float* ad = adst + (size_t)h * Dd;
    float sf = 0.f, sg = 0.f;
    for (int d = lane; d < Dd; d += 32) {
        float wv = w[d];
        sf += wv * as[d];
        sg += wv * ad[d];
    }
#pragma unroll
    for (int o = 16; o > 0; o >>= 1) {
        sf += __shfl_xor(sf, o, 32);
        sg += __shfl_xor(sg, o, 32);
    }
    if (lane == 0) { f[row] = sf; g[row] = sg; }
}

// ---------------- masked leaky-relu row softmax -> att (fp32) ----------------
__global__ __launch_bounds__(256)
void att_kernel(const float* __restrict__ f, const float* __restrict__ g,
                const float* __restrict__ adj, float* __restrict__ att, int n)
{
    int wave = threadIdx.x >> 5, lane = threadIdx.x & 31;
    int i = blockIdx.x * 8 + wave;
    if (i >= n) return;                        // wave-uniform
    float fi = f[i];
    const float* arow = adj + (size_t)i * n;
    float* orow = att + (size_t)i * n;

    float m = -3.0e38f;
    for (int j = lane; j < n; j += 32) {
        if (arow[j] > 0.f) {
            float e = fi + g[j];
            e = e >= 0.f ? e : cfg::ALPHA * e;
            m = fmaxf(m, e);
        }
    }
#pragma unroll
    for (int o = 16; o > 0; o >>= 1) m = fmaxf(m, __shfl_xor(m, o, 32));

    float s = 0.f;
    for (int j = lane; j < n; j += 32) {
        float v = 0.f;
        if (arow[j] > 0.f) {
            float e = fi + g[j];
            e = e >= 0.f ? e : cfg::ALPHA * e;
            v = __expf(e - m);
        }
        orow[j] = v;
        s += v;
    }
#pragma unroll
    for (int o = 16; o > 0; o >>= 1) s += __shfl_xor(s, o, 32);
    float inv = 1.f / s;
    for (int j = lane; j < n; j += 32) orow[j] *= inv;
}

// ---------------- gated fusion: out = z*c + (1-z)*t, z = sigmoid(c+t) ----------------
__global__ __launch_bounds__(256)
void fuse_kernel(const float* __restrict__ c, const float* __restrict__ t,
                 float* __restrict__ out, int count)
{
    int i = blockIdx.x * blockDim.x + threadIdx.x;
    if (i >= count) return;
    float cv = c[i], tv = t[i];
    float z = 1.f / (1.f + __expf(-(cv + tv)));
    out[i] = z * cv + (1.f - z) * tv;
}

// ---------------- host launcher ----------------
static inline dim3 fgrid(int M, int Nc) { return dim3(Nc / BN, M / BM, 1); }
static inline dim3 sgrid(int M, int Nc) { return dim3((Nc + BN - 1) / BN, (M + SM - 1) / SM, 1); }

extern "C" void kernel_launch(void* const* d_in, const int* in_sizes, int n_in,
                              void* d_out, int out_size, void* d_ws, size_t ws_size,
                              hipStream_t stream) {
    using namespace cfg;
    (void)in_sizes; (void)n_in; (void)out_size; (void)ws_size;

    const float* x     = (const float*)d_in[0];
    const float* adj   = (const float*)d_in[1];
    const float* t_x   = (const float*)d_in[2];
    const float* t_adj = (const float*)d_in[3];
    const float* tfidf = (const float*)d_in[4];
    const float* gW    = (const float*)d_in[5];
    const float* gasrc = (const float*)d_in[6];
    const float* gadst = (const float*)d_in[7];
    const float* gfcW  = (const float*)d_in[8];
    const float* gfcb  = (const float*)d_in[9];
    const float* tW    = (const float*)d_in[10];
    const float* tasrc = (const float*)d_in[11];
    const float* tadst = (const float*)d_in[12];
    const float* tfcW  = (const float*)d_in[13];
    const float* tfcb  = (const float*)d_in[14];
    const float* fusW  = (const float*)d_in[15];
    const float* fusb  = (const float*)d_in[16];
    const float* gc1W  = (const float*)d_in[17];
    const float* gc2W  = (const float*)d_in[18];
    const float* gc3W  = (const float*)d_in[19];

    float* out    = (float*)d_out;
    float* recon  = out;                       // [N,N]
    float* mu     = out + (size_t)N * N;       // [N,H2]
    float* logvar = mu + (size_t)N * H2;       // [N,H2]

    // workspace layout (fp32); every chunk size is a multiple of 4 floats
    float* ws = (float*)d_ws;
    float* Whc    = ws;                               ws += (size_t)NH * N * D;
    float* Wht    = ws;                               ws += (size_t)NH * T * D;
    float* f_c    = ws;                               ws += (size_t)NH * N;
    float* g_c    = ws;                               ws += (size_t)NH * N;
    float* f_t    = ws;                               ws += (size_t)NH * T;
    float* g_t    = ws;                               ws += (size_t)NH * T;
    float* att    = ws;                               ws += (size_t)N * N;        // reused per head
    float* gath_c = ws;                               ws += (size_t)N * NH * D;
    float* gath_t = ws;                               ws += (size_t)T * NH * D;
    float* conc   = ws;                               ws += (size_t)N * D;
    float* gtext  = ws;                               ws += (size_t)T * D;
    float* ctext  = ws;                               ws += (size_t)N * D;
    float* fusein = ws;                               ws += (size_t)N * D;
    float* fusion = ws;                               ws += (size_t)N * D;
    float* tmp1   = ws;                               ws += (size_t)N * H1;
    float* h1b    = ws;                               ws += (size_t)N * H1;
    float* tmp2   = ws;                               ws += (size_t)N * H2;
    float* tmp3   = ws;                               ws += (size_t)N * H2;

    // ---------- concept-graph GAT ----------
    for (int h = 0; h < NH; ++h)
        gemm_wmma_fast<false, false, 0><<<fgrid(N, D), 256, 0, stream>>>(
            x, D, gW + (size_t)h * D * D, D, Whc + (size_t)h * N * D, D, D, nullptr);
    fg_kernel<<<(NH * N + 7) / 8, 256, 0, stream>>>(Whc, gasrc, gadst, f_c, g_c, N, D, NH);
    for (int h = 0; h < NH; ++h) {
        att_kernel<<<(N + 7) / 8, 256, 0, stream>>>(f_c + (size_t)h * N, g_c + (size_t)h * N, adj, att, N);
        gemm_wmma_fast<false, false, 2><<<fgrid(N, D), 256, 0, stream>>>(   // ELU
            att, N, Whc + (size_t)h * N * D, D, gath_c + (size_t)h * D, NH * D, N, nullptr);
    }
    gemm_wmma_fast<false, false, 0><<<fgrid(N, D), 256, 0, stream>>>(
        gath_c, NH * D, gfcW, D, conc, D, NH * D, gfcb);

    // ---------- text-graph GAT ----------
    for (int h = 0; h < NH; ++h)
        gemm_wmma_fast<false, false, 0><<<fgrid(T, D), 256, 0, stream>>>(
            t_x, D, tW + (size_t)h * D * D, D, Wht + (size_t)h * T * D, D, D, nullptr);
    fg_kernel<<<(NH * T + 7) / 8, 256, 0, stream>>>(Wht, tasrc, tadst, f_t, g_t, T, D, NH);
    for (int h = 0; h < NH; ++h) {
        att_kernel<<<(T + 7) / 8, 256, 0, stream>>>(f_t + (size_t)h * T, g_t + (size_t)h * T, t_adj, att, T);
        gemm_wmma_fast<false, false, 2><<<fgrid(T, D), 256, 0, stream>>>(   // ELU
            att, T, Wht + (size_t)h * T * D, D, gath_t + (size_t)h * D, NH * D, T, nullptr);
    }
    gemm_wmma_fast<false, false, 0><<<fgrid(T, D), 256, 0, stream>>>(
        gath_t, NH * D, tfcW, D, gtext, D, NH * D, tfcb);

    // ---------- c_text = tfidf^T @ gat_text  (transposed-A fast path) ----------
    gemm_wmma_fast<true, false, 0><<<fgrid(N, D), 256, 0, stream>>>(
        tfidf, N, gtext, D, ctext, D, T, nullptr);

    // ---------- gated fusion + fusion fc ----------
    fuse_kernel<<<(N * D + 255) / 256, 256, 0, stream>>>(conc, ctext, fusein, N * D);
    gemm_wmma_fast<false, false, 0><<<fgrid(N, D), 256, 0, stream>>>(
        fusein, D, fusW, D, fusion, D, D, fusb);

    // ---------- GCN-VAE encoder ----------
    gemm_wmma_fast<false, false, 0><<<fgrid(N, H1), 256, 0, stream>>>(
        fusion, D, gc1W, H1, tmp1, H1, D, nullptr);
    gemm_wmma_fast<false, false, 1><<<fgrid(N, H1), 256, 0, stream>>>(     // ReLU
        adj, N, tmp1, H1, h1b, H1, N, nullptr);

    gemm_wmma_safe<0><<<sgrid(N, H2), 256, 0, stream>>>(
        h1b, H1, gc2W, H2, tmp2, H2, N, H2, H1, nullptr);
    gemm_wmma_safe<0><<<sgrid(N, H2), 256, 0, stream>>>(
        adj, N, tmp2, H2, mu, H2, N, H2, N, nullptr);

    gemm_wmma_safe<0><<<sgrid(N, H2), 256, 0, stream>>>(
        h1b, H1, gc3W, H2, tmp3, H2, N, H2, H1, nullptr);
    gemm_wmma_safe<0><<<sgrid(N, H2), 256, 0, stream>>>(
        adj, N, tmp3, H2, logvar, H2, N, H2, N, nullptr);

    // ---------- recon = mu @ mu^T  (transposed-B fast path, single K-tile) ----------
    gemm_wmma_fast<false, true, 0><<<fgrid(N, N), 256, 0, stream>>>(
        mu, H2, mu, H2, recon, N, H2, nullptr);
}
